// Block_55336358643145
// MI455X (gfx1250) — matrix operand order repaired and checked by reading, not compiled, not run on hardware
//
#include <hip/hip_runtime.h>
#include <hip/hip_bf16.h>

#define D_EMBD 1024
#define SEQ    2048
#define NB     2
#define NH     16
#define HD     64
#define BS     (NB * SEQ)      // 4096 rows
#define DFF    (4 * D_EMBD)    // 4096
#define JD     25

typedef __bf16 bf16_t;
typedef bf16_t v16bf __attribute__((ext_vector_type(16)));
typedef float  v8f   __attribute__((ext_vector_type(8)));

union AFrag { v16bf v; uint4 q[2]; };

__device__ __forceinline__ v8f wmma_bf16(v16bf a, v16bf b, v8f c) {
    // (neg_a, A, neg_b, B, c_mod, C, reuse_a, reuse_b)
    return __builtin_amdgcn_wmma_f32_16x16x32_bf16(false, a, false, b,
                                                   (short)0, c, false, false);
}

__device__ __forceinline__ unsigned short f2bf_u16(float x) {
    __hip_bfloat16 h = __float2bfloat16(x);
    unsigned short u;
    __builtin_memcpy(&u, &h, 2);
    return u;
}

// ---------------------------------------------------------------------------
// Transpose f32 [R][C] -> bf16 [C][R]  (weights to N-major for B fragments)
// ---------------------------------------------------------------------------
__global__ __launch_bounds__(256)
void transpose_to_bf16(const float* __restrict__ src, __hip_bfloat16* __restrict__ dst,
                       int R, int C) {
    __shared__ float tile[32][33];
    const int c0 = blockIdx.x * 32;
    const int r0 = blockIdx.y * 32;
    const int tx = threadIdx.x;
    for (int i = threadIdx.y; i < 32; i += 8)
        tile[i][tx] = src[(size_t)(r0 + i) * C + c0 + tx];
    __syncthreads();
    for (int i = threadIdx.y; i < 32; i += 8)
        dst[(size_t)(c0 + i) * R + r0 + tx] = __float2bfloat16(tile[tx][i]);
}

// ---------------------------------------------------------------------------
// LayerNorm f32 [rows][D] -> bf16 [rows][D]
// ---------------------------------------------------------------------------
__global__ __launch_bounds__(256)
void layernorm_to_bf16(const float* __restrict__ x, const float* __restrict__ g,
                       const float* __restrict__ bi, __hip_bfloat16* __restrict__ out,
                       int D) {
    __shared__ float ss[256], ss2[256];
    const int row = blockIdx.x;
    const int t = threadIdx.x;
    const float* xr = x + (size_t)row * D;
    float s = 0.f, s2 = 0.f;
    for (int i = t; i < D; i += 256) { float v = xr[i]; s += v; s2 += v * v; }
    ss[t] = s; ss2[t] = s2;
    __syncthreads();
    for (int k = 128; k > 0; k >>= 1) {
        if (t < k) { ss[t] += ss[t + k]; ss2[t] += ss2[t + k]; }
        __syncthreads();
    }
    const float mean = ss[0] / D;
    const float var  = ss2[0] / D - mean * mean;
    const float rstd = rsqrtf(var + 1e-5f);
    for (int i = t; i < D; i += 256) {
        float v = (xr[i] - mean) * rstd * g[i] + bi[i];
        out[(size_t)row * D + i] = __float2bfloat16(v);
    }
}

// ---------------------------------------------------------------------------
// Generic WMMA bf16 GEMM: C[M,N] = A[M,K] * Bt[N,K]^T (+bias, epilogue modes)
// wave tile = 64(M) x 64(N); block = 8 waves; 16 WMMAs per 16 b128 loads
// mode 0: bf16 out, (acc+bias)*scale          (q / k / generic)
// mode 1: bf16 out scattered to V^T layout     [b][h][hd][s]
// mode 2: bf16 out, exact GELU                 (MLP1)
// mode 3: f32 out, acc+bias+resid              (proj, MLP2/final)
// ---------------------------------------------------------------------------
__global__ __launch_bounds__(256)
void gemm_bf16_kernel(const __hip_bfloat16* __restrict__ A,
                      const __hip_bfloat16* __restrict__ Bt,
                      const float* __restrict__ bias,
                      int M, int N, int K, int mode, float scale,
                      __hip_bfloat16* __restrict__ outb,
                      float* __restrict__ outf,
                      const float* __restrict__ resid) {
    const int lane = threadIdx.x & 31;
    const int wave = threadIdx.x >> 5;
    const int ln = lane & 15;
    const int hi = lane >> 4;
    const int m0 = blockIdx.x * 64;
    const int n0 = (blockIdx.y * 8 + wave) * 64;
    if (n0 >= N) return;

    const __hip_bfloat16* arow[4];
#pragma unroll
    for (int i = 0; i < 4; ++i)
        arow[i] = A + (size_t)(m0 + i * 16 + ln) * K + hi * 16;
    const __hip_bfloat16* bcol[4];
#pragma unroll
    for (int j = 0; j < 4; ++j)
        bcol[j] = Bt + (size_t)(n0 + j * 16 + ln) * K + hi * 16;

    v8f acc[4][4] = {};

    for (int k = 0; k < K; k += 32) {
        AFrag a[4];
#pragma unroll
        for (int i = 0; i < 4; ++i) {
            a[i].q[0] = *(const uint4*)(arow[i] + k);
            a[i].q[1] = *(const uint4*)(arow[i] + k + 8);
        }
        v16bf bf[4];
#pragma unroll
        for (int j = 0; j < 4; ++j)
            bf[j] = *(const v16bf*)(bcol[j] + k);
#pragma unroll
        for (int j = 0; j < 4; ++j) {
#pragma unroll
            for (int i = 0; i < 4; ++i)
                acc[i][j] = wmma_bf16(a[i].v, bf[j], acc[i][j]);
        }
    }

#pragma unroll
    for (int i = 0; i < 4; ++i) {
#pragma unroll
        for (int r = 0; r < 8; ++r) {
            const int m = m0 + i * 16 + hi * 8 + r;
#pragma unroll
            for (int j = 0; j < 4; ++j) {
                const int n = n0 + j * 16 + ln;
                float val = acc[i][j][r] + bias[n];
                if (mode == 0) {
                    outb[(size_t)m * N + n] = __float2bfloat16(val * scale);
                } else if (mode == 1) {
                    const int bb = m >> 11, s = m & (SEQ - 1);
                    const int hh = n >> 6, d = n & (HD - 1);
                    outb[((size_t)((bb * NH + hh) * HD + d) << 11) + s] =
                        __float2bfloat16(val);
                } else if (mode == 2) {
                    float gg = 0.5f * val * (1.0f + erff(val * 0.70710678118654752f));
                    outb[(size_t)m * N + n] = __float2bfloat16(gg);
                } else {
                    outf[(size_t)m * N + n] = val + resid[(size_t)m * N + n];
                }
            }
        }
    }
}

// ---------------------------------------------------------------------------
// Flash attention: wave = (b, h, 16-query tile); streams 32-key tiles.
// q,k row-major bf16 [b*s][D] (q pre-scaled 1/sqrt(hd)); v transposed
// bf16 [b][h][hd][s]. Causal mask + zeroed every (JD-1 mod JD) key column.
// ---------------------------------------------------------------------------
__global__ __launch_bounds__(256)
void attention_kernel(const __hip_bfloat16* __restrict__ qb,
                      const __hip_bfloat16* __restrict__ kb,
                      const __hip_bfloat16* __restrict__ vt,
                      __hip_bfloat16* __restrict__ ob) {
    __shared__ __align__(16) unsigned short plds[8][16][40];
    const int lane = threadIdx.x & 31;
    const int wave = threadIdx.x >> 5;
    const int ln = lane & 15;
    const int hi = lane >> 4;
    const int unit = blockIdx.x * 8 + wave;
    const int qt = unit & (SEQ / 16 - 1);
    const int bh = unit >> 7;
    const int b = bh >> 4;
    const int h = bh & (NH - 1);
    const int s0 = qt * 16;

    // q A-fragments for the two 32-wide K slices of hd=64
    const __hip_bfloat16* qrow = qb + (size_t)(b * SEQ + s0 + ln) * D_EMBD + h * HD;
    AFrag aq[2];
#pragma unroll
    for (int kk = 0; kk < 2; ++kk) {
        aq[kk].q[0] = *(const uint4*)(qrow + kk * 32 + hi * 16);
        aq[kk].q[1] = *(const uint4*)(qrow + kk * 32 + hi * 16 + 8);
    }

    v8f o[4] = {};
    float mrow[8], lrow[8];
#pragma unroll
    for (int r = 0; r < 8; ++r) { mrow[r] = -1e30f; lrow[r] = 0.f; }

    const int ntiles = (s0 + 16 + 31) >> 5;
    for (int nt = 0; nt < ntiles; ++nt) {
        const int kbase = nt * 32;
        v8f sc[2] = {};
#pragma unroll
        for (int f = 0; f < 2; ++f) {
            const int key = kbase + f * 16 + ln;
            const __hip_bfloat16* kp =
                kb + (size_t)(b * SEQ + key) * D_EMBD + h * HD + hi * 16;
            v16bf b0 = *(const v16bf*)(kp);
            v16bf b1 = *(const v16bf*)(kp + 32);
            sc[f] = wmma_bf16(aq[0].v, b0, sc[f]);
            sc[f] = wmma_bf16(aq[1].v, b1, sc[f]);
        }

        // mask + online softmax (rows striped across 16-lane halves)
#pragma unroll
        for (int r = 0; r < 8; ++r) {
            const int srow = s0 + hi * 8 + r;
            const int key0 = kbase + ln;
            const int key1 = kbase + 16 + ln;
            float v0 = sc[0][r], v1 = sc[1][r];
            if (!((key0 <= srow) && (key0 % JD != JD - 1))) v0 = -1e30f;
            if (!((key1 <= srow) && (key1 % JD != JD - 1))) v1 = -1e30f;
            float mx = fmaxf(v0, v1);
            mx = fmaxf(mx, __shfl_xor(mx, 1, 32));
            mx = fmaxf(mx, __shfl_xor(mx, 2, 32));
            mx = fmaxf(mx, __shfl_xor(mx, 4, 32));
            mx = fmaxf(mx, __shfl_xor(mx, 8, 32));
            const float mnew = fmaxf(mrow[r], mx);
            const float alpha = __expf(mrow[r] - mnew);
            mrow[r] = mnew;
            const float p0 = __expf(v0 - mnew);
            const float p1 = __expf(v1 - mnew);
            float rs = p0 + p1;
            rs += __shfl_xor(rs, 1, 32);
            rs += __shfl_xor(rs, 2, 32);
            rs += __shfl_xor(rs, 4, 32);
            rs += __shfl_xor(rs, 8, 32);
            lrow[r] = lrow[r] * alpha + rs;
#pragma unroll
            for (int j = 0; j < 4; ++j) o[j][r] *= alpha;
            const int prow = hi * 8 + r;
            plds[wave][prow][ln] = f2bf_u16(p0);
            plds[wave][prow][16 + ln] = f2bf_u16(p1);
        }
        asm volatile("s_wait_dscnt 0" ::: "memory");

        // re-shape P (C layout) -> A fragment via LDS
        AFrag ap;
        const unsigned short* lp = &plds[wave][ln][0];
        ap.q[0] = *(const uint4*)(lp + hi * 16);
        ap.q[1] = *(const uint4*)(lp + hi * 16 + 8);

#pragma unroll
        for (int j = 0; j < 4; ++j) {
            const __hip_bfloat16* vp =
                vt + (size_t)(bh * HD + j * 16 + ln) * SEQ + kbase + hi * 16;
            v16bf vf = *(const v16bf*)vp;
            o[j] = wmma_bf16(ap.v, vf, o[j]);
        }
    }

#pragma unroll
    for (int r = 0; r < 8; ++r) {
        const float inv = 1.0f / lrow[r];
        const int srow = s0 + hi * 8 + r;
        __hip_bfloat16* op = ob + (size_t)(b * SEQ + srow) * D_EMBD + h * HD;
#pragma unroll
        for (int j = 0; j < 4; ++j)
            op[j * 16 + ln] = __float2bfloat16(o[j][r] * inv);
    }
}

// ---------------------------------------------------------------------------
extern "C" void kernel_launch(void* const* d_in, const int* in_sizes, int n_in,
                              void* d_out, int out_size, void* d_ws, size_t ws_size,
                              hipStream_t stream) {
    const float* hidden = (const float*)d_in[0];
    const float* Wq = (const float*)d_in[1];
    const float* bq = (const float*)d_in[2];
    const float* Wk = (const float*)d_in[3];
    const float* bk = (const float*)d_in[4];
    const float* Wv = (const float*)d_in[5];
    const float* bv = (const float*)d_in[6];
    const float* Wp = (const float*)d_in[7];
    const float* bp = (const float*)d_in[8];
    const float* ln1g = (const float*)d_in[9];
    const float* ln1b = (const float*)d_in[10];
    const float* ln2g = (const float*)d_in[11];
    const float* ln2b = (const float*)d_in[12];
    const float* W1 = (const float*)d_in[13];
    const float* b1 = (const float*)d_in[14];
    const float* W2 = (const float*)d_in[15];
    const float* b2 = (const float*)d_in[16];
    float* out = (float*)d_out;

    char* ws = (char*)d_ws;
    const size_t MB = 1024ull * 1024ull;
    __hip_bfloat16* WqT   = (__hip_bfloat16*)(ws + 0 * MB);    // 2 MB
    __hip_bfloat16* WkT   = (__hip_bfloat16*)(ws + 2 * MB);    // 2 MB
    __hip_bfloat16* WvT   = (__hip_bfloat16*)(ws + 4 * MB);    // 2 MB
    __hip_bfloat16* WpT   = (__hip_bfloat16*)(ws + 6 * MB);    // 2 MB
    __hip_bfloat16* W1T   = (__hip_bfloat16*)(ws + 8 * MB);    // 8 MB
    __hip_bfloat16* W2T   = (__hip_bfloat16*)(ws + 16 * MB);   // 8 MB
    __hip_bfloat16* xb    = (__hip_bfloat16*)(ws + 24 * MB);   // 8 MB
    __hip_bfloat16* qbuf  = (__hip_bfloat16*)(ws + 32 * MB);   // 8 MB
    __hip_bfloat16* kbuf  = (__hip_bfloat16*)(ws + 40 * MB);   // 8 MB
    __hip_bfloat16* vtb   = (__hip_bfloat16*)(ws + 48 * MB);   // 8 MB
    __hip_bfloat16* attnb = (__hip_bfloat16*)(ws + 56 * MB);   // 8 MB
    float*          hbuf  = (float*)(ws + 64 * MB);            // 16 MB
    __hip_bfloat16* h2b   = (__hip_bfloat16*)(ws + 80 * MB);   // 8 MB
    __hip_bfloat16* m1    = (__hip_bfloat16*)(ws + 88 * MB);   // 32 MB -> 120 MB total

    const dim3 tb(32, 8);
    // weight transposes f32 -> bf16 [N][K]
    transpose_to_bf16<<<dim3(32, 32), tb, 0, stream>>>(Wq, WqT, D_EMBD, D_EMBD);
    transpose_to_bf16<<<dim3(32, 32), tb, 0, stream>>>(Wk, WkT, D_EMBD, D_EMBD);
    transpose_to_bf16<<<dim3(32, 32), tb, 0, stream>>>(Wv, WvT, D_EMBD, D_EMBD);
    transpose_to_bf16<<<dim3(32, 32), tb, 0, stream>>>(Wp, WpT, D_EMBD, D_EMBD);
    transpose_to_bf16<<<dim3(128, 32), tb, 0, stream>>>(W1, W1T, D_EMBD, DFF);
    transpose_to_bf16<<<dim3(32, 128), tb, 0, stream>>>(W2, W2T, DFF, D_EMBD);

    // LN1
    layernorm_to_bf16<<<BS, 256, 0, stream>>>(hidden, ln1g, ln1b, xb, D_EMBD);

    // QKV (q pre-scaled by 1/sqrt(hd)=0.125; v stored transposed per head)
    const dim3 gq(BS / 64, D_EMBD / 512);
    gemm_bf16_kernel<<<gq, 256, 0, stream>>>(xb, WqT, bq, BS, D_EMBD, D_EMBD,
                                             0, 0.125f, qbuf, nullptr, nullptr);
    gemm_bf16_kernel<<<gq, 256, 0, stream>>>(xb, WkT, bk, BS, D_EMBD, D_EMBD,
                                             0, 1.0f, kbuf, nullptr, nullptr);
    gemm_bf16_kernel<<<gq, 256, 0, stream>>>(xb, WvT, bv, BS, D_EMBD, D_EMBD,
                                             1, 1.0f, vtb, nullptr, nullptr);

    // attention: 2*16*(2048/16) = 4096 wave-units / 8 waves per block
    attention_kernel<<<512, 256, 0, stream>>>(qbuf, kbuf, vtb, attnb);

    // projection + residual -> hbuf (f32)
    gemm_bf16_kernel<<<gq, 256, 0, stream>>>(attnb, WpT, bp, BS, D_EMBD, D_EMBD,
                                             3, 1.0f, nullptr, hbuf, hidden);

    // LN2
    layernorm_to_bf16<<<BS, 256, 0, stream>>>(hbuf, ln2g, ln2b, h2b, D_EMBD);

    // MLP1 (GELU) and MLP2 (+residual2) -> out
    gemm_bf16_kernel<<<dim3(BS / 64, DFF / 512), 256, 0, stream>>>(
        h2b, W1T, b1, BS, DFF, D_EMBD, 2, 1.0f, m1, nullptr, nullptr);
    gemm_bf16_kernel<<<gq, 256, 0, stream>>>(m1, W2T, b2, BS, D_EMBD, DFF,
                                             3, 1.0f, nullptr, out, hbuf);
}
